// SlidingWindowAttention_76682346103107
// MI455X (gfx1250) — matrix operand order repaired and checked by reading, not compiled, not run on hardware
//
#include <hip/hip_runtime.h>
#include <cstdint>
#include <cstddef>

// ---------------------------------------------------------------------------
// Types for CDNA5 WMMA / TDM (gfx1250, wave32)
// ---------------------------------------------------------------------------
typedef __attribute__((ext_vector_type(8)))  __bf16 v8bf;
typedef __attribute__((ext_vector_type(16))) __bf16 v16bf;
typedef __attribute__((ext_vector_type(8)))  float  v8f;
typedef unsigned int u32x4 __attribute__((ext_vector_type(4)));
typedef int          i32x8 __attribute__((ext_vector_type(8)));
typedef int          i32x4 __attribute__((ext_vector_type(4)));

static __device__ __forceinline__ v16bf cat16(v8bf lo, v8bf hi) {
  return __builtin_shufflevector(lo, hi, 0,1,2,3,4,5,6,7,8,9,10,11,12,13,14,15);
}

static __device__ __forceinline__ v8f wmma_bf16(v16bf a, v16bf b, v8f c) {
  // D = A(16x32 bf16) * B(32x16 bf16) + C(16x16 f32)
  return __builtin_amdgcn_wmma_f32_16x16x32_bf16(false, a, false, b, (short)0, c,
                                                 false, false);
}

// ---------------------------------------------------------------------------
// Problem constants
// ---------------------------------------------------------------------------
#define S_LEN   4096
#define DMODEL  2048
#define NH      32
#define NKV     8
#define HEADD   64
#define WND     512
#define INTERN  5632
#define QKVN    3072   // 2048 q + 512 k + 512 v
#define LN10000 9.210340371976184f

// ---------------------------------------------------------------------------
// f32 -> bf16 pack (with column offset/stride for qkv weight packing)
// ---------------------------------------------------------------------------
__global__ void pack_bf16_kernel(const float* __restrict__ src,
                                 __bf16* __restrict__ dst,
                                 int cols, int dstStride, int dstOff, int total) {
  int idx = blockIdx.x * 256 + threadIdx.x;
  if (idx >= total) return;
  int r = idx / cols;
  int c = idx - r * cols;
  dst[(size_t)r * dstStride + dstOff + c] = (__bf16)src[idx];
}

// ---------------------------------------------------------------------------
// RMSNorm: out_bf16 = x * rsqrt(mean(x^2)+eps) * w * scale   (row per block)
// ---------------------------------------------------------------------------
__global__ __launch_bounds__(256) void rmsnorm_kernel(const float* __restrict__ x,
                                                      const float* __restrict__ w,
                                                      __bf16* __restrict__ out,
                                                      float scale) {
  __shared__ float red[256];
  const int row = blockIdx.x;
  const int tid = threadIdx.x;
  const float* xr = x + (size_t)row * DMODEL;
  float s = 0.f;
#pragma unroll
  for (int i = 0; i < DMODEL / 256; ++i) {
    float v = xr[tid + i * 256];
    s += v * v;
  }
  red[tid] = s;
  __syncthreads();
  for (int off = 128; off > 0; off >>= 1) {
    if (tid < off) red[tid] += red[tid + off];
    __syncthreads();
  }
  float r = rsqrtf(red[0] * (1.0f / DMODEL) + 1e-5f) * scale;
#pragma unroll
  for (int i = 0; i < DMODEL / 256; ++i) {
    int j = tid + i * 256;
    out[(size_t)row * DMODEL + j] = (__bf16)(xr[j] * r * w[j]);
  }
}

// ---------------------------------------------------------------------------
// Generic bf16 WMMA GEMM: C[M,N] = A[M,K] @ B[K,N]
//   block: 256 threads (8 waves), tile 128x128, K-step 32
//   A tiles streamed global->LDS by the Tensor Data Mover (TDM), double
//   buffered, one tile in flight (per-wave 16-row slices, TENSORcnt-tracked).
//   TDM pad_enable reproduces the 40-element LDS row stride in hardware.
// MODE 0: Cf = acc
// MODE 1: Cf = Rf + acc*scale           (residual fuse)
// MODE 2: Cb = bf16(silu(acc))
// MODE 3: Cb = bf16(acc * float(Xb))    (gate * y fuse)
// ---------------------------------------------------------------------------
template <int MODE>
__global__ __launch_bounds__(256) void gemm_bf16_kernel(
    const __bf16* __restrict__ A, const __bf16* __restrict__ B, int ldb,
    float* __restrict__ Cf, __bf16* __restrict__ Cb,
    const float* __restrict__ Rf, const __bf16* __restrict__ Xb,
    float scale, int M, int N, int K) {
  __shared__ __bf16 As[2][128 * 40];   // [row][k]   stride 40 elems (80B)
  __shared__ __bf16 Bt[2][128 * 40];   // [col][k]   transposed B, stride 40

  const int tid  = threadIdx.x;
  const int lane = tid & 31;
  const int w    = tid >> 5;
  const int wm   = w >> 1;         // 0..3
  const int wn   = w & 1;          // 0..1
  const int l15  = lane & 15;
  const int hi   = lane >> 4;      // 0 or 1
  const int row0 = blockIdx.y * 128;
  const int col0 = blockIdx.x * 128;

  // --- per-wave TDM descriptor (16 rows x 32 cols bf16, 2D) ---
  const int wv = __builtin_amdgcn_readfirstlane(w);       // force SGPR
  const __bf16* aSlice = A + (size_t)(row0 + wv * 16) * K; // wave's row slice
  const unsigned int ldsA0 =
      (unsigned int)(uintptr_t)(&As[0][0]) + (unsigned int)(wv * 16 * 80);
  const unsigned int ldsA1 =
      (unsigned int)(uintptr_t)(&As[1][0]) + (unsigned int)(wv * 16 * 80);
  i32x8 g1;
  g1[0] = (int)((1u << 16) |      // data_size = 2 bytes
                (1u << 20) |      // pad_enable
                (3u << 22) |      // pad_interval: 16 DWORDs (one 64B row)
                (3u << 25));      // pad_amount:   4 DWORDs (16B) -> stride 80B
  g1[1] = (int)(((unsigned)K & 0xFFFFu) << 16);               // tensor_dim0 lo
  g1[2] = (int)(((unsigned)K >> 16) | (((unsigned)M & 0xFFFFu) << 16));
  g1[3] = (int)(((unsigned)M >> 16) | (32u << 16));           // tile_dim0 = 32
  g1[4] = 16;                                                 // tile_dim1 = 16
  g1[5] = (int)(unsigned)K;                                   // dim0_stride
  g1[6] = 0;
  g1[7] = 0;
  i32x4 gz4;
  gz4[0] = 0; gz4[1] = 0; gz4[2] = 0; gz4[3] = 0;
  i32x8 gz8;
  gz8[0] = 0; gz8[1] = 0; gz8[2] = 0; gz8[3] = 0;
  gz8[4] = 0; gz8[5] = 0; gz8[6] = 0; gz8[7] = 0;

  auto issueA = [&](int k0, int buf) {
    unsigned long long ga = (unsigned long long)(uintptr_t)(aSlice + k0);
    u32x4 g0;
    g0[0] = 1u;                                         // count=1, user D#
    g0[1] = buf ? ldsA1 : ldsA0;                        // lds_addr
    g0[2] = (unsigned int)ga;                           // global_addr lo
    g0[3] = ((unsigned int)(ga >> 32) & 0x01FFFFFFu) | (2u << 30); // type=2
    __builtin_amdgcn_tensor_load_to_lds(g0, g1, gz4, gz4, gz8, 0);
  };

  v8f zero = {};
  v8f acc[2][4];
#pragma unroll
  for (int mi = 0; mi < 2; ++mi)
#pragma unroll
    for (int ni = 0; ni < 4; ++ni) acc[mi][ni] = zero;

  const int nK = K / 32;
  issueA(0, 0);                      // prologue: tile 0 in flight

  for (int i = 0; i < nK; ++i) {
    const int k0  = i * 32;
    const int buf = i & 1;
    // ---- stage B tile 32x128 manually, write transposed [col][k] ----
#pragma unroll
    for (int it = 0; it < 2; ++it) {
      int idx = tid + it * 256;          // 512 chunks of 8 bf16
      int kk = idx >> 4, nc = idx & 15;
      v8bf vb = *(const v8bf*)(B + (size_t)(k0 + kk) * ldb + col0 + nc * 8);
#pragma unroll
      for (int j = 0; j < 8; ++j) Bt[buf][(nc * 8 + j) * 40 + kk] = vb[j];
    }
    // ---- issue next A tile into the other buffer, wait for current ----
    if (i + 1 < nK) {
      issueA(k0 + 32, buf ^ 1);
      __builtin_prefetch(B + (size_t)(k0 + 32 + (tid >> 4)) * ldb + col0, 0, 1);
      __builtin_amdgcn_s_wait_tensorcnt(1);   // in-order: tile i complete
    } else {
      __builtin_amdgcn_s_wait_tensorcnt(0);
    }
    __syncthreads();

    // ---- fragments per ISA 16-bit A 16x32 / B 32x16 layouts ----
    v16bf af[2];
#pragma unroll
    for (int mi = 0; mi < 2; ++mi) {
      int r = wm * 32 + mi * 16 + l15;
      int kb = hi * 8;
      af[mi] = cat16(*(const v8bf*)(&As[buf][r * 40 + kb]),
                     *(const v8bf*)(&As[buf][r * 40 + 16 + kb]));
    }
    v16bf bfr[4];
#pragma unroll
    for (int ni = 0; ni < 4; ++ni) {
      int n = wn * 64 + ni * 16 + l15;
      int kk = hi * 8;
      bfr[ni] = cat16(*(const v8bf*)(&Bt[buf][n * 40 + kk]),
                      *(const v8bf*)(&Bt[buf][n * 40 + kk + 16]));
    }
#pragma unroll
    for (int mi = 0; mi < 2; ++mi)
#pragma unroll
      for (int ni = 0; ni < 4; ++ni)
        acc[mi][ni] = wmma_bf16(af[mi], bfr[ni], acc[mi][ni]);
    __syncthreads();   // protect buffers before TDM/B-staging reuse
  }

  // ---- epilogue: C layout -> row m = r + 8*hi, col n = l15 ----
#pragma unroll
  for (int mi = 0; mi < 2; ++mi)
#pragma unroll
    for (int ni = 0; ni < 4; ++ni)
#pragma unroll
      for (int r = 0; r < 8; ++r) {
        int m = row0 + wm * 32 + mi * 16 + r + 8 * hi;
        int n = col0 + wn * 64 + ni * 16 + l15;
        float v = acc[mi][ni][r];
        size_t o = (size_t)m * N + n;
        if (MODE == 0) {
          Cf[o] = v;
        } else if (MODE == 1) {
          Cf[o] = Rf[o] + v * scale;
        } else if (MODE == 2) {
          Cb[o] = (__bf16)(v / (1.f + __expf(-v)));
        } else {
          Cb[o] = (__bf16)(v * (float)Xb[o]);
        }
      }
}

// ---------------------------------------------------------------------------
// RoPE for Q: qkv f32 [S,3072] (q at col 0) -> qr bf16 [H][S][64]
// ---------------------------------------------------------------------------
__global__ void rope_q_kernel(const float* __restrict__ qkv,
                              __bf16* __restrict__ qr) {
  int idx = blockIdx.x * 256 + threadIdx.x;   // S*H*32 threads
  int d = idx & 31;
  int t = idx >> 5;
  int h = t & (NH - 1);
  int s = t >> 5;
  float freq = (float)s * __expf(-(float)d * (LN10000 / 32.f));
  float sn, c;
  __sincosf(freq, &sn, &c);
  const float* base = qkv + (size_t)s * QKVN + h * HEADD;
  float x1 = base[d], x2 = base[d + 32];
  size_t o = (size_t)h * S_LEN * HEADD + (size_t)s * HEADD + d;
  qr[o]      = (__bf16)(x1 * c - x2 * sn);
  qr[o + 32] = (__bf16)(x2 * c + x1 * sn);
}

// ---------------------------------------------------------------------------
// RoPE for K + transpose V: kr bf16 [KVH][S][64], vt bf16 [KVH][64][S]
// ---------------------------------------------------------------------------
__global__ void rope_kv_kernel(const float* __restrict__ qkv,
                               __bf16* __restrict__ kr,
                               __bf16* __restrict__ vt) {
  int idx = blockIdx.x * 256 + threadIdx.x;   // S*KVH*32 threads
  int d = idx & 31;
  int t = idx >> 5;
  int h = t & (NKV - 1);
  int s = t >> 3;
  float freq = (float)s * __expf(-(float)d * (LN10000 / 32.f));
  float sn, c;
  __sincosf(freq, &sn, &c);
  const float* kb = qkv + (size_t)s * QKVN + DMODEL + h * HEADD;
  float x1 = kb[d], x2 = kb[d + 32];
  size_t o = (size_t)h * S_LEN * HEADD + (size_t)s * HEADD + d;
  kr[o]      = (__bf16)(x1 * c - x2 * sn);
  kr[o + 32] = (__bf16)(x2 * c + x1 * sn);
  const float* vb = qkv + (size_t)s * QKVN + (DMODEL + NKV * HEADD) + h * HEADD;
  vt[((size_t)h * HEADD + d) * S_LEN + s]      = (__bf16)vb[d];
  vt[((size_t)h * HEADD + d + 32) * S_LEN + s] = (__bf16)vb[d + 32];
}

// ---------------------------------------------------------------------------
// Sliding-window attention, flash-style. 1 wave per (head, 32-query tile).
//   qr [H][S][64], kr [KVH][S][64], vt [KVH][64][S] -> attn bf16 [S][H*64]
// ---------------------------------------------------------------------------
__global__ __launch_bounds__(32) void attn_kernel(const __bf16* __restrict__ qr,
                                                  const __bf16* __restrict__ kr,
                                                  const __bf16* __restrict__ vt,
                                                  __bf16* __restrict__ attn) {
  __shared__ __bf16 Plds[32 * 40];   // P transpose staging, stride 40

  const int lane = threadIdx.x;
  const int l15  = lane & 15;
  const int hi   = lane >> 4;
  const int head = blockIdx.y;
  const int kvh  = head >> 2;        // rep = H/KVH = 4
  const int qb   = blockIdx.x * 32;

  const __bf16* qh = qr + (size_t)head * S_LEN * HEADD;
  const __bf16* kh = kr + (size_t)kvh * S_LEN * HEADD;
  const __bf16* vh = vt + (size_t)kvh * HEADD * S_LEN;

  // Q fragments (persistent): 2 row-halves x 2 hd-chunks of 32
  v16bf aq[2][2];
#pragma unroll
  for (int mi = 0; mi < 2; ++mi)
#pragma unroll
    for (int kj = 0; kj < 2; ++kj) {
      int row = qb + mi * 16 + l15;
      int base = kj * 32 + hi * 8;
      aq[mi][kj] = cat16(*(const v8bf*)(qh + (size_t)row * HEADD + base),
                         *(const v8bf*)(qh + (size_t)row * HEADD + base + 16));
    }

  v8f zero = {};
  v8f out[2][4];
#pragma unroll
  for (int mi = 0; mi < 2; ++mi)
#pragma unroll
    for (int nj = 0; nj < 4; ++nj) out[mi][nj] = zero;
  float mrow[16], lrow[16];
#pragma unroll
  for (int i = 0; i < 16; ++i) { mrow[i] = -1e30f; lrow[i] = 0.f; }

  int kcs = qb - (WND - 1);
  if (kcs < 0) kcs = 0;
  kcs &= ~31;

  for (int kc = kcs; kc <= qb; kc += 32) {
    // K^T fragments direct from global (contiguous hd per lane)
    v16bf bk[2][2];
#pragma unroll
    for (int ni = 0; ni < 2; ++ni)
#pragma unroll
      for (int kj = 0; kj < 2; ++kj) {
        int key = kc + ni * 16 + l15;
        int hd0 = kj * 32 + hi * 8;
        bk[ni][kj] = cat16(*(const v8bf*)(kh + (size_t)key * HEADD + hd0),
                           *(const v8bf*)(kh + (size_t)key * HEADD + hd0 + 16));
      }
    // scores: chain 2 WMMAs over HD=64
    v8f sc[2][2];
#pragma unroll
    for (int mi = 0; mi < 2; ++mi)
#pragma unroll
      for (int ni = 0; ni < 2; ++ni) {
        v8f t = wmma_bf16(aq[mi][0], bk[ni][0], zero);
        sc[mi][ni] = wmma_bf16(aq[mi][1], bk[ni][1], t);
      }
    // mask + scale (q,k already carry PRE_SCALE each -> matches reference)
    float sv[2][2][8];
#pragma unroll
    for (int mi = 0; mi < 2; ++mi)
#pragma unroll
      for (int ni = 0; ni < 2; ++ni)
#pragma unroll
        for (int r = 0; r < 8; ++r) {
          int m = qb + mi * 16 + r + 8 * hi;
          int kpos = kc + ni * 16 + l15;
          float v = sc[mi][ni][r] * 0.125f;   // HD^-0.5
          bool ok = (kpos <= m) && (m - kpos < WND);
          sv[mi][ni][r] = ok ? v : -1e30f;
        }
    // online softmax per row (16-lane butterfly within each half)
#pragma unroll
    for (int mi = 0; mi < 2; ++mi)
#pragma unroll
      for (int r = 0; r < 8; ++r) {
        int idx = mi * 8 + r;
        float rm = fmaxf(sv[mi][0][r], sv[mi][1][r]);
        for (int off = 1; off < 16; off <<= 1)
          rm = fmaxf(rm, __shfl_xor(rm, off, 32));
        float mnew  = fmaxf(mrow[idx], rm);
        float alpha = __expf(mrow[idx] - mnew);
        mrow[idx] = mnew;
        float p0 = __expf(sv[mi][0][r] - mnew);
        float p1 = __expf(sv[mi][1][r] - mnew);
        sv[mi][0][r] = p0;
        sv[mi][1][r] = p1;
        float rs = p0 + p1;
        for (int off = 1; off < 16; off <<= 1)
          rs += __shfl_xor(rs, off, 32);
        lrow[idx] = lrow[idx] * alpha + rs;
#pragma unroll
        for (int nj = 0; nj < 4; ++nj) out[mi][nj][r] *= alpha;
      }
    // transpose P through LDS into A-fragments
    __syncthreads();
#pragma unroll
    for (int mi = 0; mi < 2; ++mi)
#pragma unroll
      for (int ni = 0; ni < 2; ++ni)
#pragma unroll
        for (int r = 0; r < 8; ++r)
          Plds[(mi * 16 + r + 8 * hi) * 40 + ni * 16 + l15] =
              (__bf16)sv[mi][ni][r];
    __syncthreads();

    v16bf ap[2];
#pragma unroll
    for (int mi = 0; mi < 2; ++mi) {
      int row = mi * 16 + l15;
      int kb = hi * 8;
      ap[mi] = cat16(*(const v8bf*)(&Plds[row * 40 + kb]),
                     *(const v8bf*)(&Plds[row * 40 + 16 + kb]));
    }
    // V fragments direct from transposed V (contiguous keys per lane)
    v16bf bv[4];
#pragma unroll
    for (int nj = 0; nj < 4; ++nj) {
      int hd = nj * 16 + l15;
      const __bf16* p = vh + (size_t)hd * S_LEN + kc + hi * 8;
      bv[nj] = cat16(*(const v8bf*)(p), *(const v8bf*)(p + 16));
    }
#pragma unroll
    for (int mi = 0; mi < 2; ++mi)
#pragma unroll
      for (int nj = 0; nj < 4; ++nj)
        out[mi][nj] = wmma_bf16(ap[mi], bv[nj], out[mi][nj]);
  }

  // normalize and write [s][h*64+hd]
#pragma unroll
  for (int mi = 0; mi < 2; ++mi)
#pragma unroll
    for (int nj = 0; nj < 4; ++nj)
#pragma unroll
      for (int r = 0; r < 8; ++r) {
        int m = qb + mi * 16 + r + 8 * hi;
        int hd = nj * 16 + l15;
        float denom = lrow[mi * 8 + r];
        attn[(size_t)m * DMODEL + head * HEADD + hd] =
            (__bf16)(out[mi][nj][r] / denom);
      }
}

// ---------------------------------------------------------------------------
// Launcher
// ---------------------------------------------------------------------------
extern "C" void kernel_launch(void* const* d_in, const int* in_sizes, int n_in,
                              void* d_out, int out_size, void* d_ws, size_t ws_size,
                              hipStream_t stream) {
  (void)in_sizes; (void)n_in; (void)out_size; (void)ws_size;
  const float* x     = (const float*)d_in[0];   // [S, D]
  const float* mixw  = (const float*)d_in[1];   // [D]
  const float* wq    = (const float*)d_in[2];   // [D, 2048]
  const float* wk    = (const float*)d_in[3];   // [D, 512]
  const float* wv    = (const float*)d_in[4];   // [D, 512]
  const float* wo    = (const float*)d_in[5];   // [2048, D]
  const float* mlpw  = (const float*)d_in[6];   // [D]
  const float* wg    = (const float*)d_in[7];   // [D, 2*INTER]
  const float* wd    = (const float*)d_in[8];   // [INTER, D]
  float* out = (float*)d_out;                   // [S, D]

  char* ws = (char*)d_ws;
  size_t off = 0;
  auto alloc = [&](size_t bytes) {
    size_t o = off;
    off = (off + bytes + 255) & ~(size_t)255;
    return o;
  };
  __bf16* wqkv_b = (__bf16*)(ws + alloc((size_t)DMODEL * QKVN * 2));
  __bf16* wo_b   = (__bf16*)(ws + alloc((size_t)DMODEL * DMODEL * 2));
  __bf16* wg_b   = (__bf16*)(ws + alloc((size_t)DMODEL * 2 * INTERN * 2));
  __bf16* wd_b   = (__bf16*)(ws + alloc((size_t)INTERN * DMODEL * 2));
  __bf16* xn1    = (__bf16*)(ws + alloc((size_t)S_LEN * DMODEL * 2));
  float*  qkv    = (float*)(ws + alloc((size_t)S_LEN * QKVN * 4));
  __bf16* q_r    = (__bf16*)(ws + alloc((size_t)NH * S_LEN * HEADD * 2));
  __bf16* k_r    = (__bf16*)(ws + alloc((size_t)NKV * S_LEN * HEADD * 2));
  __bf16* v_t    = (__bf16*)(ws + alloc((size_t)NKV * HEADD * S_LEN * 2));
  __bf16* attnb  = (__bf16*)(ws + alloc((size_t)S_LEN * DMODEL * 2));
  float*  hbuf   = (float*)(ws + alloc((size_t)S_LEN * DMODEL * 4));
  __bf16* xn2    = (__bf16*)(ws + alloc((size_t)S_LEN * DMODEL * 2));
  __bf16* sgate  = (__bf16*)(ws + alloc((size_t)S_LEN * INTERN * 2));
  __bf16* actb   = (__bf16*)(ws + alloc((size_t)S_LEN * INTERN * 2));

  const float PRE_SCALE = 0.70710678118654752f;   // 1/sqrt(2)
  const float INV_PRE   = 1.41421356237309505f;   // sqrt(2)

  // 1) pack weights to bf16 (wq|wk|wv fused into one [D,3072] matrix)
  {
    int t;
    t = DMODEL * DMODEL;
    pack_bf16_kernel<<<(t + 255) / 256, 256, 0, stream>>>(wq, wqkv_b, DMODEL, QKVN, 0, t);
    t = DMODEL * 512;
    pack_bf16_kernel<<<(t + 255) / 256, 256, 0, stream>>>(wk, wqkv_b, 512, QKVN, 2048, t);
    pack_bf16_kernel<<<(t + 255) / 256, 256, 0, stream>>>(wv, wqkv_b, 512, QKVN, 2560, t);
    t = DMODEL * DMODEL;
    pack_bf16_kernel<<<(t + 255) / 256, 256, 0, stream>>>(wo, wo_b, DMODEL, DMODEL, 0, t);
    t = DMODEL * 2 * INTERN;
    pack_bf16_kernel<<<(t + 255) / 256, 256, 0, stream>>>(wg, wg_b, 2 * INTERN, 2 * INTERN, 0, t);
    t = INTERN * DMODEL;
    pack_bf16_kernel<<<(t + 255) / 256, 256, 0, stream>>>(wd, wd_b, DMODEL, DMODEL, 0, t);
  }

  // 2) rmsnorm1 (includes PRE_SCALE)
  rmsnorm_kernel<<<S_LEN, 256, 0, stream>>>(x, mixw, xn1, PRE_SCALE);

  // 3) fused QKV GEMM: [S,2048] @ [2048,3072] -> f32
  gemm_bf16_kernel<0><<<dim3(QKVN / 128, S_LEN / 128), 256, 0, stream>>>(
      xn1, wqkv_b, QKVN, qkv, nullptr, nullptr, nullptr, 0.f,
      S_LEN, QKVN, DMODEL);

  // 4) RoPE + head-major / V-transpose repack
  rope_q_kernel<<<(S_LEN * NH * 32) / 256, 256, 0, stream>>>(qkv, q_r);
  rope_kv_kernel<<<(S_LEN * NKV * 32) / 256, 256, 0, stream>>>(qkv, k_r, v_t);

  // 5) sliding-window attention
  attn_kernel<<<dim3(S_LEN / 32, NH), 32, 0, stream>>>(q_r, k_r, v_t, attnb);

  // 6) output proj, fused residual: h = x + (attn @ wo) / PRE_SCALE
  gemm_bf16_kernel<1><<<dim3(DMODEL / 128, S_LEN / 128), 256, 0, stream>>>(
      attnb, wo_b, DMODEL, hbuf, nullptr, x, nullptr, INV_PRE,
      S_LEN, DMODEL, DMODEL);

  // 7) rmsnorm2
  rmsnorm_kernel<<<S_LEN, 256, 0, stream>>>(hbuf, mlpw, xn2, 1.0f);

  // 8) gate half with fused SiLU -> bf16
  gemm_bf16_kernel<2><<<dim3(INTERN / 128, S_LEN / 128), 256, 0, stream>>>(
      xn2, wg_b, 2 * INTERN, nullptr, sgate, nullptr, nullptr, 0.f,
      S_LEN, INTERN, DMODEL);

  // 9) y half with fused multiply by silu(gate) -> bf16
  gemm_bf16_kernel<3><<<dim3(INTERN / 128, S_LEN / 128), 256, 0, stream>>>(
      xn2, wg_b + INTERN, 2 * INTERN, nullptr, actb, nullptr, sgate, 0.f,
      S_LEN, INTERN, DMODEL);

  // 10) down proj, fused residual: out = h + act @ w_down
  gemm_bf16_kernel<1><<<dim3(DMODEL / 128, S_LEN / 128), 256, 0, stream>>>(
      actb, wd_b, DMODEL, out, nullptr, hbuf, nullptr, 1.0f,
      S_LEN, DMODEL, INTERN);
}
